// DiffAttention_15977278341927
// MI455X (gfx1250) — matrix operand order repaired and checked by reading, not compiled, not run on hardware
//
#include <hip/hip_runtime.h>
#include <hip/hip_bf16.h>

#define TOK  4096   // B*N
#define DIMK 1024
#define NSEQ 2048
#define NH   16
#define NE   8
#define HD   64

typedef __attribute__((ext_vector_type(16))) __bf16 v16bf;
typedef __attribute__((ext_vector_type(8)))  __bf16 v8bf;
typedef __attribute__((ext_vector_type(8)))  float  v8f;

union BF16x16 { v16bf v; v8bf h[2]; };

// A-fragment (16x32 bf16, M x K): lane L: M=L%16, half h=L/16.
// elements 0..7  -> K = 16h + e       (VGPRs 0..3)
// elements 8..15 -> K = 16h + e       (VGPRs 4..7)
__device__ __forceinline__ v16bf load_afrag(const __bf16* row, int lh) {
  BF16x16 u;
  u.h[0] = *(const v8bf*)(row + 16 * lh);
  u.h[1] = *(const v8bf*)(row + 16 * lh + 8);
  return u.v;
}

// LDS byte offset of a shared-memory object: generic pointers to LDS are
// {shared_aperture_hi32, lds_offset}, so the low 32 bits are the DS address.
__device__ __forceinline__ unsigned lds_off(const void* p) {
  return (unsigned)(unsigned long long)p;
}

// ---------------------------------------------------------------------------
// Conversion / transpose kernels
// ---------------------------------------------------------------------------
__global__ void f32_to_bf16(const float* __restrict__ in, __bf16* __restrict__ out, int n) {
  int i = blockIdx.x * blockDim.x + threadIdx.x;
  for (; i < n; i += gridDim.x * blockDim.x) out[i] = (__bf16)in[i];
}

__global__ void transpose_cvt(const float* __restrict__ W, __bf16* __restrict__ WT) {
  int i = blockIdx.x * blockDim.x + threadIdx.x;   // 1M threads, 1024x1024
  int k = i >> 10, n = i & 1023;
  WT[(size_t)n * DIMK + k] = (__bf16)W[(size_t)k * DIMK + n];
}

// ---------------------------------------------------------------------------
// WMMA GEMM: C(4096x1024) = Xbf(4096x1024) @ W + bias, W given transposed bf16.
// One wave computes a 16x64 strip (4 column tiles), K-loop step 32.
// MODE 0: Q  -> *0.125, store bf16 (b,h,n,64)
// MODE 1: K  -> store bf16 (b,h,n,64)
// MODE 2: V  -> store bf16 transposed (b,e,128,n)
// MODE 3: out-> store f32 row-major (final projection)
// ---------------------------------------------------------------------------
template <int MODE>
__global__ __launch_bounds__(256)
void gemm16x64(const __bf16* __restrict__ X, const __bf16* __restrict__ WT,
               const float* __restrict__ bias,
               __bf16* __restrict__ outb, float* __restrict__ outf)
{
  const int lane = threadIdx.x & 31;
  const int lm = lane & 15, lh = lane >> 4;
  const int wid = blockIdx.x * 8 + (threadIdx.x >> 5);
  const int rb = wid >> 4;    // 0..255 : 16-row block
  const int cs = wid & 15;    // 0..15  : 64-col strip

  v8f acc[4] = {};
  const __bf16* xrow = X + (size_t)(rb * 16 + lm) * DIMK;

#pragma unroll 1
  for (int k0 = 0; k0 < DIMK; k0 += 32) {
    if (k0 + 128 < DIMK) {
      __builtin_prefetch(xrow + k0 + 128, 0, 1);                      // global_prefetch_b8
      __builtin_prefetch(WT + (size_t)(cs * 64 + lm) * DIMK + k0 + 128, 0, 1);
    }
    v16bf a = load_afrag(xrow + k0, lh);
#pragma unroll
    for (int t = 0; t < 4; ++t) {
      const int n = cs * 64 + t * 16 + lm;
      // B-fragment (32x16, K x N): lane = column n, elements = K = 16h + e
      v16bf b = *(const v16bf*)(WT + (size_t)n * DIMK + k0 + 16 * lh);
      acc[t] = __builtin_amdgcn_wmma_f32_16x16x32_bf16(
          false, a, false, b, (short)0, acc[t], false, false);
    }
  }

  const int row0 = rb * 16 + 8 * lh;   // C/D layout: element j -> row row0+j
#pragma unroll
  for (int t = 0; t < 4; ++t) {
    const int c = cs * 64 + t * 16 + lm;
    const float bc = bias[c];
    if constexpr (MODE == 0 || MODE == 1) {
      const int h = c >> 6, d = c & 63;
#pragma unroll
      for (int j = 0; j < 8; ++j) {
        const int row = row0 + j;
        const int bb = row >> 11, n = row & 2047;
        float v = acc[t][j] + bc;
        if constexpr (MODE == 0) v *= 0.125f;   // HEAD_DIM^-0.5
        outb[(((size_t)(bb * NH + h)) * NSEQ + n) * HD + d] = (__bf16)v;
      }
    } else if constexpr (MODE == 2) {
      const int e = c >> 7, d = c & 127;
      const int bb = row0 >> 11, n0 = row0 & 2047;   // 8 consecutive tokens
      v8bf pk;
#pragma unroll
      for (int j = 0; j < 8; ++j) pk[j] = (__bf16)(acc[t][j] + bc);
      *(v8bf*)(outb + (((size_t)(bb * NE + e)) * 128 + d) * NSEQ + n0) = pk;
    } else {
#pragma unroll
      for (int j = 0; j < 8; ++j)
        outf[(size_t)(row0 + j) * DIMK + c] = acc[t][j] + bc;
    }
  }
}

// ---------------------------------------------------------------------------
// Differential attention.
// Block = 8 waves, all on the same (b, eff-head); wave w owns 16-query block
// rb = (blockIdx&15)*8 + w. Per 32-key step the block async-copies the K tiles
// of BOTH sub-heads (2x32x64 bf16) and the V tile (128x32 bf16) into LDS once,
// shared by all 8 waves (GLOBAL_LOAD_ASYNC_TO_LDS_B128 + s_wait_asynccnt).
// Row-sums of P come from an extra WMMA against an all-ones B fragment.
// ---------------------------------------------------------------------------
__global__ __launch_bounds__(256)
void diffattn(const __bf16* __restrict__ Qb, const __bf16* __restrict__ Kb,
              const __bf16* __restrict__ Vt, const float* __restrict__ g,
              const float* __restrict__ lq1, const float* __restrict__ lk1,
              const float* __restrict__ lq2, const float* __restrict__ lk2,
              __bf16* __restrict__ Ab)
{
  __shared__ __bf16 Ks[2 * 32 * HD];   // [head][key][d]   8 KB
  __shared__ __bf16 Vs[128 * 32];      // [d][key]         8 KB
  __shared__ __bf16 Ps[8 * 16 * 32];   // per-wave P tile  8 KB

  const int tid  = threadIdx.x;
  const int lane = tid & 31;
  const int lm = lane & 15, lh = lane >> 4;
  const int wv = tid >> 5;
  const int pair = blockIdx.x >> 4;          // 0..15 = (b,e)
  const int b = pair >> 3, e = pair & 7;
  const int rb = (blockIdx.x & 15) * 8 + wv; // 0..127 query block
  __bf16* pl = Ps + wv * 512;

  float s1 = 0.f, s2 = 0.f;
  for (int i = 0; i < 64; ++i) { s1 += lq1[i] * lk1[i]; s2 += lq2[i] * lk2[i]; }
  const float lam = __expf(s1) - __expf(s2) + 0.8f;

  v16bf onesf;
#pragma unroll
  for (int i = 0; i < 16; ++i) onesf[i] = (__bf16)1.0f;

  // Q fragments for both sub-heads (reused over all key blocks)
  v16bf qa[2][2];
#pragma unroll
  for (int hh = 0; hh < 2; ++hh) {
    const __bf16* qrow =
        Qb + (((size_t)(b * NH + 2 * e + hh)) * NSEQ + rb * 16 + lm) * HD;
    qa[hh][0] = load_afrag(qrow, lh);
    qa[hh][1] = load_afrag(qrow + 32, lh);
  }

  const __bf16* Kbase[2] = {
      Kb + ((size_t)(b * NH + 2 * e)) * NSEQ * HD,
      Kb + ((size_t)(b * NH + 2 * e + 1)) * NSEQ * HD };
  const __bf16* Vbase = Vt + ((size_t)pair) * 128 * NSEQ;

  float mr[2][8], lr[2][8];
  v8f acc[2][8] = {};
#pragma unroll
  for (int hh = 0; hh < 2; ++hh)
#pragma unroll
    for (int j = 0; j < 8; ++j) { mr[hh][j] = -3.0e38f; lr[hh][j] = 0.f; }

#pragma unroll 1
  for (int kb = 0; kb < NSEQ; kb += 32) {
    __syncthreads();   // all waves done reading the previous tile
    // ---- async-stage K (both heads) and V tiles into LDS: 4 x b128/thread --
#pragma unroll
    for (int r = 0; r < 2; ++r) {          // K: i = head*256 + key*8 + seg
      const int i = tid + r * 256;
      const int key = (i >> 3) & 31, seg = i & 7;   // head == r (uniform)
      const unsigned voff = (unsigned)(((kb + key) * HD + seg * 8) * 2);
      const unsigned dst  = lds_off(&Ks[(r * 32 + key) * HD + seg * 8]);
      asm volatile("global_load_async_to_lds_b128 %0, %1, %2"
                   :: "v"(dst), "v"(voff),
                      "s"((unsigned long long)Kbase[r]) : "memory");
    }
#pragma unroll
    for (int r = 0; r < 2; ++r) {          // V: d = r*64 + tid/4, seg = tid&3
      const int d = r * 64 + (tid >> 2), seg = tid & 3;
      const unsigned voff = (unsigned)((d * NSEQ + kb + seg * 8) * 2);
      const unsigned dst  = lds_off(&Vs[d * 32 + seg * 8]);
      asm volatile("global_load_async_to_lds_b128 %0, %1, %2"
                   :: "v"(dst), "v"(voff),
                      "s"((unsigned long long)Vbase) : "memory");
    }
    asm volatile("s_wait_asynccnt 0" ::: "memory");
    __syncthreads();   // staged tile visible to all waves

#pragma unroll
    for (int hh = 0; hh < 2; ++hh) {
      // ---- S = Q @ K^T for 32 keys (two 16x16 tiles) from LDS ----
      v8f sf[2];
#pragma unroll
      for (int s = 0; s < 2; ++s) {
        v8f c = {};
        const __bf16* kr = Ks + (hh * 32 + s * 16 + lm) * HD + 16 * lh;
        v16bf k0 = *(const v16bf*)(kr);
        v16bf k1 = *(const v16bf*)(kr + 32);
        c = __builtin_amdgcn_wmma_f32_16x16x32_bf16(false, qa[hh][0], false, k0, (short)0, c, false, false);
        c = __builtin_amdgcn_wmma_f32_16x16x32_bf16(false, qa[hh][1], false, k1, (short)0, c, false, false);
        sf[s] = c;
      }
      // ---- row max across the 32 keys (lanes hold columns) ----
      float bm[8];
#pragma unroll
      for (int j = 0; j < 8; ++j) bm[j] = fmaxf(sf[0][j], sf[1][j]);
#pragma unroll
      for (int off = 1; off < 16; off <<= 1)
#pragma unroll
        for (int j = 0; j < 8; ++j) bm[j] = fmaxf(bm[j], __shfl_xor(bm[j], off, 32));
      float sc[8];
#pragma unroll
      for (int j = 0; j < 8; ++j) {
        const float mn = fmaxf(mr[hh][j], bm[j]);
        sc[j] = __expf(mr[hh][j] - mn);
        mr[hh][j] = mn;
      }
      // ---- P = exp(S - m), relayout C/D -> A through per-wave LDS tile ----
#pragma unroll
      for (int s = 0; s < 2; ++s)
#pragma unroll
        for (int j = 0; j < 8; ++j)
          pl[(8 * lh + j) * 32 + 16 * s + lm] = (__bf16)__expf(sf[s][j] - mr[hh][j]);
      v16bf pa = load_afrag(pl + lm * 32, lh);   // DS in-order within the wave

      // ---- row sums via WMMA with all-ones B (replaces shuffle reduce) ----
      v8f zero = {};
      v8f rsum = __builtin_amdgcn_wmma_f32_16x16x32_bf16(
          false, pa, false, onesf, (short)0, zero, false, false);
#pragma unroll
      for (int j = 0; j < 8; ++j) lr[hh][j] = lr[hh][j] * sc[j] + rsum[j];
#pragma unroll
      for (int f = 0; f < 8; ++f)
#pragma unroll
        for (int j = 0; j < 8; ++j) acc[hh][f][j] *= sc[j];

      // ---- acc += P @ V from the shared LDS V tile ----
#pragma unroll
      for (int f = 0; f < 8; ++f) {
        v16bf vb = *(const v16bf*)(Vs + (16 * f + lm) * 32 + 16 * lh);
        acc[hh][f] = __builtin_amdgcn_wmma_f32_16x16x32_bf16(
            false, pa, false, vb, (short)0, acc[hh][f], false, false);
      }
    }
  }

  // ---- diff, RMSNorm over 128 dims, *g*0.2, store activation ----
  float ss[8];
#pragma unroll
  for (int j = 0; j < 8; ++j) ss[j] = 0.f;
  v8f fin[8];
#pragma unroll
  for (int f = 0; f < 8; ++f)
#pragma unroll
    for (int j = 0; j < 8; ++j) {
      const float v = acc[0][f][j] / lr[0][j] - lam * (acc[1][f][j] / lr[1][j]);
      fin[f][j] = v;
      ss[j] += v * v;
    }
#pragma unroll
  for (int off = 1; off < 16; off <<= 1)
#pragma unroll
    for (int j = 0; j < 8; ++j) ss[j] += __shfl_xor(ss[j], off, 32);
  float rms[8];
#pragma unroll
  for (int j = 0; j < 8; ++j) rms[j] = rsqrtf(ss[j] * (1.f / 128.f) + 1e-5f);

#pragma unroll
  for (int f = 0; f < 8; ++f) {
    const float gw = g[16 * f + lm];
#pragma unroll
    for (int j = 0; j < 8; ++j) {
      const int row = rb * 16 + 8 * lh + j;
      const float v = fin[f][j] * rms[j] * gw * 0.2f;
      Ab[((size_t)b * NSEQ + row) * DIMK + e * 128 + 16 * f + lm] = (__bf16)v;
    }
  }
}

// ---------------------------------------------------------------------------
extern "C" void kernel_launch(void* const* d_in, const int* in_sizes, int n_in,
                              void* d_out, int out_size, void* d_ws, size_t ws_size,
                              hipStream_t stream) {
  const float* x   = (const float*)d_in[0];
  const float* Wq  = (const float*)d_in[1];
  const float* bq  = (const float*)d_in[2];
  const float* Wk  = (const float*)d_in[3];
  const float* bk  = (const float*)d_in[4];
  const float* Wv  = (const float*)d_in[5];
  const float* bv  = (const float*)d_in[6];
  const float* Wo  = (const float*)d_in[7];
  const float* bo  = (const float*)d_in[8];
  const float* g   = (const float*)d_in[9];
  const float* lq1 = (const float*)d_in[10];
  const float* lk1 = (const float*)d_in[11];
  const float* lq2 = (const float*)d_in[12];
  const float* lk2 = (const float*)d_in[13];
  float* out = (float*)d_out;

  char* ws = (char*)d_ws;                       // 48 MB used
  __bf16* Xb  = (__bf16*)(ws);                  //  8 MB  x bf16
  __bf16* WqT = (__bf16*)(ws + (8u  << 20));    //  2 MB  each, transposed bf16
  __bf16* WkT = (__bf16*)(ws + (10u << 20));
  __bf16* WvT = (__bf16*)(ws + (12u << 20));
  __bf16* WoT = (__bf16*)(ws + (14u << 20));
  __bf16* Qb  = (__bf16*)(ws + (16u << 20));    //  8 MB (b,h,n,64)
  __bf16* Kb  = (__bf16*)(ws + (24u << 20));    //  8 MB (b,h,n,64)
  __bf16* Vt  = (__bf16*)(ws + (32u << 20));    //  8 MB (b,e,128,n)
  __bf16* Ab  = (__bf16*)(ws + (40u << 20));    //  8 MB (b,n,1024)

  f32_to_bf16<<<4096, 256, 0, stream>>>(x, Xb, TOK * DIMK);
  transpose_cvt<<<4096, 256, 0, stream>>>(Wq, WqT);
  transpose_cvt<<<4096, 256, 0, stream>>>(Wk, WkT);
  transpose_cvt<<<4096, 256, 0, stream>>>(Wv, WvT);
  transpose_cvt<<<4096, 256, 0, stream>>>(Wo, WoT);

  gemm16x64<0><<<512, 256, 0, stream>>>(Xb, WqT, bq, Qb, nullptr);
  gemm16x64<1><<<512, 256, 0, stream>>>(Xb, WkT, bk, Kb, nullptr);
  gemm16x64<2><<<512, 256, 0, stream>>>(Xb, WvT, bv, Vt, nullptr);

  diffattn<<<256, 256, 0, stream>>>(Qb, Kb, Vt, g, lq1, lk1, lq2, lk2, Ab);

  gemm16x64<3><<<512, 256, 0, stream>>>(Ab, WoT, bo, nullptr, out);
}
